// PointnetSAModuleMSG_SSD_28097676050928
// MI455X (gfx1250) — matrix operand order repaired and checked by reading, not compiled, not run on hardware
//
#include <hip/hip_runtime.h>
#include <hip/hip_bf16.h>

#define BATCH 2
#define NPTS  16384
#define CIN   64
#define MPT   4096
#define COLS  64   // columns per branch block

typedef __attribute__((ext_vector_type(16))) __bf16 v16bf;
typedef __attribute__((ext_vector_type(8)))  float  v8f;

__device__ __forceinline__ __bf16 us2bf(unsigned short u) {
    return __builtin_bit_cast(__bf16, u);
}
__device__ __forceinline__ unsigned short f2bf(float f) {
    unsigned int x = __float_as_uint(f);
    unsigned int r = (x + 0x7FFFu + ((x >> 16) & 1u)) >> 16;
    return (unsigned short)r;
}

// A fragment (16x32 bf16, rows of W, row-major, stride in elems): contiguous k = 16*grp + e.
__device__ __forceinline__ v16bf load_A_frag(const unsigned short* W, int row0, int k0, int stride) {
    const int lane = threadIdx.x & 31;
    const int grp  = lane >> 4;
    const int mrow = lane & 15;
    v16bf a;
#pragma unroll
    for (int e = 0; e < 16; ++e) {
        const int v = e >> 1, odd = e & 1;
        const int k = grp * 16 + e;
        a[e] = us2bf(W[(row0 + mrow) * stride + k0 + k]);
    }
    return a;
}
// B fragment from K-major (transposed) activations Xt[col][K]: contiguous 16 elems/lane.
__device__ __forceinline__ v16bf load_B_frag_T(const unsigned short* Xt, int k0, int col0, int stride) {
    const int lane = threadIdx.x & 31;
    const int grp  = lane >> 4;
    const int n    = lane & 15;
    const unsigned short* p = Xt + (col0 + n) * stride + k0 + grp * 16;
    v16bf b;
#pragma unroll
    for (int e = 0; e < 16; ++e) b[e] = us2bf(p[e]);
    return b;
}
// B fragment from row-major X[K][cols].
__device__ __forceinline__ v16bf load_B_frag(const unsigned short* X, int k0, int col0, int stride) {
    const int lane = threadIdx.x & 31;
    const int grp  = lane >> 4;
    const int n    = lane & 15;
    v16bf b;
#pragma unroll
    for (int e = 0; e < 16; ++e) b[e] = us2bf(X[(k0 + grp * 16 + e) * stride + col0 + n]);
    return b;
}

// ---------------------------------------------------------------------------
// Kernel 0: one-shot f32 -> bf16 weight conversion with K zero-padding.
// ---------------------------------------------------------------------------
__global__ void pad_convert_kernel(const float* __restrict__ src, unsigned short* __restrict__ dst,
                                   int rows, int ksrc, int kdst) {
    const int total = rows * kdst;
    for (int i = blockIdx.x * blockDim.x + threadIdx.x; i < total; i += gridDim.x * blockDim.x) {
        const int r = i / kdst, k = i % kdst;
        dst[i] = (k < ksrc) ? f2bf(src[r * ksrc + k]) : (unsigned short)0;
    }
}

// ---------------------------------------------------------------------------
// Kernel 1: farthest point sampling. grid = BATCH, block = 1024 (32 waves).
// ---------------------------------------------------------------------------
__global__ void fps_kernel(const float* __restrict__ xyz, float* __restrict__ new_xyz) {
    const int b    = blockIdx.x;
    const int tid  = threadIdx.x;
    const int lane = tid & 31;
    const int wv   = tid >> 5;

    __shared__ float s_last[3];
    __shared__ float s_rv[32];
    __shared__ int   s_ri[32];

    float px[16], py[16], pz[16], d[16];
#pragma unroll
    for (int i = 0; i < 16; ++i) {
        const int p = tid + i * 1024;
        px[i] = xyz[((size_t)b * NPTS + p) * 3 + 0];
        py[i] = xyz[((size_t)b * NPTS + p) * 3 + 1];
        pz[i] = xyz[((size_t)b * NPTS + p) * 3 + 2];
        d[i]  = 1e10f;
    }
    if (tid == 0) {
        s_last[0] = xyz[(size_t)b * NPTS * 3 + 0];
        s_last[1] = xyz[(size_t)b * NPTS * 3 + 1];
        s_last[2] = xyz[(size_t)b * NPTS * 3 + 2];
        new_xyz[((size_t)b * MPT) * 3 + 0] = s_last[0];
        new_xyz[((size_t)b * MPT) * 3 + 1] = s_last[1];
        new_xyz[((size_t)b * MPT) * 3 + 2] = s_last[2];
    }
    __syncthreads();

    for (int m = 1; m < MPT; ++m) {
        const float lx = s_last[0], ly = s_last[1], lz = s_last[2];
        float bestv = -1.0f;
        int   besti = 0;
#pragma unroll
        for (int i = 0; i < 16; ++i) {
            const float dx = px[i] - lx, dy = py[i] - ly, dz = pz[i] - lz;
            const float dd = dx * dx + dy * dy + dz * dz;
            d[i] = fminf(d[i], dd);
            if (d[i] > bestv) { bestv = d[i]; besti = tid + i * 1024; }
        }
#pragma unroll
        for (int off = 16; off > 0; off >>= 1) {
            const float v2 = __shfl_xor(bestv, off, 32);
            const int   i2 = __shfl_xor(besti, off, 32);
            if (v2 > bestv || (v2 == bestv && i2 < besti)) { bestv = v2; besti = i2; }
        }
        if (lane == 0) { s_rv[wv] = bestv; s_ri[wv] = besti; }
        __syncthreads();
        if (wv == 0) {
            float v = s_rv[lane];
            int   i = s_ri[lane];
#pragma unroll
            for (int off = 16; off > 0; off >>= 1) {
                const float v2 = __shfl_xor(v, off, 32);
                const int   i2 = __shfl_xor(i, off, 32);
                if (v2 > v || (v2 == v && i2 < i)) { v = v2; i = i2; }
            }
            if (lane == 0) {
                const float wx = xyz[((size_t)b * NPTS + i) * 3 + 0];
                const float wy = xyz[((size_t)b * NPTS + i) * 3 + 1];
                const float wz = xyz[((size_t)b * NPTS + i) * 3 + 2];
                s_last[0] = wx; s_last[1] = wy; s_last[2] = wz;
                new_xyz[((size_t)b * MPT + m) * 3 + 0] = wx;
                new_xyz[((size_t)b * MPT + m) * 3 + 1] = wy;
                new_xyz[((size_t)b * MPT + m) * 3 + 2] = wz;
            }
        }
        __syncthreads();
    }
}

// ---------------------------------------------------------------------------
// Kernel 2: ball query for all 3 radii, one wave per (b,m) row.
// ---------------------------------------------------------------------------
__global__ void ball_query_kernel(const float* __restrict__ xyz,
                                  const float* __restrict__ new_xyz,
                                  int* __restrict__ idx1, int* __restrict__ idx2,
                                  int* __restrict__ idx3) {
    const int lane = threadIdx.x & 31;
    const int row  = blockIdx.x * 8 + (threadIdx.x >> 5);
    const int b    = row / MPT;
    const float cx = new_xyz[(size_t)row * 3 + 0];
    const float cy = new_xyz[(size_t)row * 3 + 1];
    const float cz = new_xyz[(size_t)row * 3 + 2];

    int* p1 = idx1 + (size_t)row * 32;
    int* p2 = idx2 + (size_t)row * 32;
    int* p3 = idx3 + (size_t)row * 64;

    int cnt1 = 0, cnt2 = 0, cnt3 = 0;
    int first1 = 0, first2 = 0, first3 = 0;
    const unsigned int ltmask = (1u << lane) - 1u;

    for (int j0 = 0; j0 < NPTS; j0 += 32) {
        const int j = j0 + lane;
        __builtin_prefetch(&xyz[((size_t)b * NPTS + j0 + 256) * 3], 0, 3);
        const float dx = xyz[((size_t)b * NPTS + j) * 3 + 0] - cx;
        const float dy = xyz[((size_t)b * NPTS + j) * 3 + 1] - cy;
        const float dz = xyz[((size_t)b * NPTS + j) * 3 + 2] - cz;
        const float d2 = dx * dx + dy * dy + dz * dz;

        const unsigned int m1 = (unsigned int)__ballot(d2 < 0.25f);
        const unsigned int m2 = (unsigned int)__ballot(d2 < 1.0f);
        const unsigned int m3 = (unsigned int)__ballot(d2 < 4.0f);

        if (cnt1 < 32 && m1) {
            if (cnt1 == 0) first1 = j0 + __ffs((int)m1) - 1;
            const int slot = cnt1 + __popc(m1 & ltmask);
            if (((m1 >> lane) & 1u) && slot < 32) p1[slot] = j;
            cnt1 = min(32, cnt1 + __popc(m1));
        }
        if (cnt2 < 32 && m2) {
            if (cnt2 == 0) first2 = j0 + __ffs((int)m2) - 1;
            const int slot = cnt2 + __popc(m2 & ltmask);
            if (((m2 >> lane) & 1u) && slot < 32) p2[slot] = j;
            cnt2 = min(32, cnt2 + __popc(m2));
        }
        if (cnt3 < 64 && m3) {
            if (cnt3 == 0) first3 = j0 + __ffs((int)m3) - 1;
            const int slot = cnt3 + __popc(m3 & ltmask);
            if (((m3 >> lane) & 1u) && slot < 64) p3[slot] = j;
            cnt3 = min(64, cnt3 + __popc(m3));
        }
        if (cnt1 >= 32 && cnt2 >= 32 && cnt3 >= 64) break;
    }
    for (int s = cnt1 + lane; s < 32; s += 32) p1[s] = first1;
    for (int s = cnt2 + lane; s < 32; s += 32) p2[s] = first2;
    for (int s = cnt3 + lane; s < 64; s += 32) p3[s] = first3;
}

// ---------------------------------------------------------------------------
// Kernel 3: fused group + 2-layer MLP (WMMA bf16) + max-pool for one branch.
// Weights arrive pre-converted/padded bf16. Activations staged K-major
// (transposed) so each B fragment is 2 contiguous 16B LDS reads per lane.
// ---------------------------------------------------------------------------
__global__ void branch_kernel(const float* __restrict__ xyz,
                              const float* __restrict__ feats,
                              const float* __restrict__ new_xyz,
                              const int* __restrict__ idx,
                              int ns, int C1,
                              const unsigned short* __restrict__ wb1,
                              const float* __restrict__ g1, const float* __restrict__ b1,
                              const unsigned short* __restrict__ wb2,
                              const float* __restrict__ g2, const float* __restrict__ b2,
                              unsigned short* __restrict__ cat, int cat_off) {
    __shared__ __align__(16) unsigned short sW[128 * 96];    // W1 then (aliased) W2
    __shared__ __align__(16) unsigned short sX1[COLS * 96];  // transposed: [col][K]
    __shared__ __align__(16) unsigned short sX2[COLS * 96];  // transposed: [col][K]
    __shared__ float sG1[96], sB1[96], sG2[128], sB2[128];
    __shared__ int   sPool[128 * 2];

    const int tid  = threadIdx.x;
    const int wv   = tid >> 5;
    const int lane = tid & 31;
    const int grp  = lane >> 4;
    const int nn   = lane & 15;
    const int mPerBlock = COLS / ns;

    // -------- phase 0: stage W1 (uint4 copies), constants, gather X1^T --------
    {
        const uint4* s = (const uint4*)wb1;
        uint4* dw = (uint4*)sW;
        for (int i = tid; i < C1 * 12; i += 256) dw[i] = s[i];   // C1 x 96 bf16
    }
    for (int i = tid; i < C1; i += 256) { sG1[i] = g1[i]; sB1[i] = b1[i]; }
    for (int i = tid; i < 128; i += 256) { sG2[i] = g2[i]; sB2[i] = b2[i]; }
    for (int i = tid; i < 128 * 2; i += 256) sPool[i] = 0;
    // zero-pad X2^T rows k = C1..95 for every column
    for (int i = tid; i < COLS * (96 - C1); i += 256) {
        const int c = i / (96 - C1), k = C1 + i % (96 - C1);
        sX2[c * 96 + k] = 0;
    }
    // gather grouped input, K-major
    for (int i = tid; i < COLS * 96; i += 256) {
        const int c = i / 96, k = i % 96;
        const int cg = blockIdx.x * COLS + c;
        const int b  = cg / (MPT * ns);
        const int mm = (cg % (MPT * ns)) / ns;
        const int s  = cg % ns;
        float v = 0.0f;
        if (k < 67) {
            const int id = idx[((size_t)(b * MPT + mm)) * ns + s];
            if (k < 3) {
                v = xyz[((size_t)b * NPTS + id) * 3 + k] -
                    new_xyz[((size_t)b * MPT + mm) * 3 + k];
            } else {
                v = feats[((size_t)b * CIN + (k - 3)) * NPTS + id];
            }
        }
        sX1[i] = f2bf(v);
    }
    __syncthreads();

    // -------- phase 1: layer 1 GEMM (C1 x 96) @ (96 x 64) --------
    const int tiles1 = (C1 / 16) * (COLS / 16);
    for (int t = wv; t < tiles1; t += 8) {
        const int rt = t / (COLS / 16), ct = t % (COLS / 16);
        v8f acc = {};
#pragma unroll
        for (int kt = 0; kt < 3; ++kt) {
            const v16bf a  = load_A_frag(sW, rt * 16, kt * 32, 96);
            const v16bf bm = load_B_frag_T(sX1, kt * 32, ct * 16, 96);
            acc = __builtin_amdgcn_wmma_f32_16x16x32_bf16(false, a, false, bm,
                                                          (short)0, acc, false, false);
        }
#pragma unroll
        for (int v = 0; v < 8; ++v) {
            const int r = rt * 16 + v + 8 * grp;
            const int c = ct * 16 + nn;
            const float y = fmaxf(sG1[r] * acc[v] + sB1[r], 0.0f);
            sX2[c * 96 + r] = f2bf(y);
        }
    }
    __syncthreads();

    // -------- phase 2: stage W2 (128 x 96 bf16) into aliased sW --------
    {
        const uint4* s = (const uint4*)wb2;
        uint4* dw = (uint4*)sW;
        for (int i = tid; i < 128 * 12; i += 256) dw[i] = s[i];
    }
    __syncthreads();

    // -------- phase 3: layer 2 GEMM (128 x 96) @ (96 x 64), fused max-pool --------
    for (int t = wv; t < 8 * (COLS / 16); t += 8) {
        const int rt = t / (COLS / 16), ct = t % (COLS / 16);
        v8f acc = {};
#pragma unroll
        for (int kt = 0; kt < 3; ++kt) {
            const v16bf a  = load_A_frag(sW, rt * 16, kt * 32, 96);
            const v16bf bm = load_B_frag_T(sX2, kt * 32, ct * 16, 96);
            acc = __builtin_amdgcn_wmma_f32_16x16x32_bf16(false, a, false, bm,
                                                          (short)0, acc, false, false);
        }
#pragma unroll
        for (int v = 0; v < 8; ++v) {
            const int r = rt * 16 + v + 8 * grp;
            const int c = ct * 16 + nn;
            const int ml = c / ns;
            const float y = fmaxf(sG2[r] * acc[v] + sB2[r], 0.0f);
            atomicMax(&sPool[r * 2 + ml], __float_as_int(y)); // y >= 0: int order == float order
        }
    }
    __syncthreads();

    // -------- phase 4: emit pooled (128, mPerBlock) features to cat (bf16) --------
    for (int i = tid; i < 128 * mPerBlock; i += 256) {
        const int ch = i / mPerBlock, ml = i % mPerBlock;
        const int cg = blockIdx.x * COLS + ml * ns;
        const int b  = cg / (MPT * ns);
        const int mm = (cg % (MPT * ns)) / ns;
        const float pv = __int_as_float(sPool[ch * 2 + ml]);
        cat[((size_t)b * 384 + cat_off + ch) * MPT + mm] = f2bf(pv);
    }
}

// ---------------------------------------------------------------------------
// Kernel 4: final GEMM y = relu(g*(W_out(256x384) @ cat(384,B*M)) + b).
// Block = 256 threads (8 waves) handles 256 rows x 64 cols; X tile staged in
// LDS with coalesced uint4 copies; each A fragment reused across 4 col tiles.
// ---------------------------------------------------------------------------
__global__ void out_gemm_kernel(const unsigned short* __restrict__ cat,
                                const unsigned short* __restrict__ wb,
                                const float* __restrict__ g,
                                const float* __restrict__ bb,
                                float* __restrict__ y) {
    __shared__ __align__(16) unsigned short sX[384 * 64];   // [K][64] row-major
    __shared__ float sG[256], sB[256];

    const int tid  = threadIdx.x;
    const int wv   = tid >> 5;
    const int lane = tid & 31;
    const int grp  = lane >> 4;
    const int nn   = lane & 15;
    const int mrow = lane & 15;
    const int col0g = blockIdx.x * 64;
    const int b    = col0g / MPT;
    const int m0   = col0g % MPT;

    {
        uint4* dst = (uint4*)sX;
        for (int i = tid; i < 384 * 8; i += 256) {
            const int k = i / 8, q = i % 8;
            const uint4* src = (const uint4*)(cat + ((size_t)b * 384 + k) * MPT + m0);
            dst[i] = src[q];
        }
    }
    for (int i = tid; i < 256; i += 256) { sG[i] = g[i]; sB[i] = bb[i]; }
    __syncthreads();

    for (int rr = 0; rr < 2; ++rr) {
        const int rt   = wv * 2 + rr;
        const int row0 = rt * 16;
        v8f acc[4] = {};
#pragma unroll
        for (int kt = 0; kt < 12; ++kt) {
            const int k0 = kt * 32;
            v16bf a;
#pragma unroll
            for (int e = 0; e < 16; ++e) {
                const int v = e >> 1, odd = e & 1;
                const int k = grp * 16 + e;
                a[e] = us2bf(wb[(size_t)(row0 + mrow) * 384 + k0 + k]);
            }
#pragma unroll
            for (int ct = 0; ct < 4; ++ct) {
                const v16bf bm = load_B_frag(sX, k0, ct * 16, 64);
                acc[ct] = __builtin_amdgcn_wmma_f32_16x16x32_bf16(false, a, false, bm,
                                                                  (short)0, acc[ct], false, false);
            }
        }
#pragma unroll
        for (int ct = 0; ct < 4; ++ct) {
#pragma unroll
            for (int v = 0; v < 8; ++v) {
                const int r = row0 + v + 8 * grp;
                const int m = m0 + ct * 16 + nn;
                y[((size_t)b * 256 + r) * MPT + m] = fmaxf(sG[r] * acc[ct][v] + sB[r], 0.0f);
            }
        }
    }
}

// ---------------------------------------------------------------------------
extern "C" void kernel_launch(void* const* d_in, const int* in_sizes, int n_in,
                              void* d_out, int out_size, void* d_ws, size_t ws_size,
                              hipStream_t stream) {
    const float* xyz   = (const float*)d_in[0];
    const float* feats = (const float*)d_in[1];
    const float* w11 = (const float*)d_in[2];
    const float* g11 = (const float*)d_in[3];
    const float* b11 = (const float*)d_in[4];
    const float* w12 = (const float*)d_in[5];
    const float* g12 = (const float*)d_in[6];
    const float* b12 = (const float*)d_in[7];
    const float* w21 = (const float*)d_in[8];
    const float* g21 = (const float*)d_in[9];
    const float* b21 = (const float*)d_in[10];
    const float* w22 = (const float*)d_in[11];
    const float* g22 = (const float*)d_in[12];
    const float* b22 = (const float*)d_in[13];
    const float* w31 = (const float*)d_in[14];
    const float* g31 = (const float*)d_in[15];
    const float* b31 = (const float*)d_in[16];
    const float* w32 = (const float*)d_in[17];
    const float* g32 = (const float*)d_in[18];
    const float* b32 = (const float*)d_in[19];
    const float* wo  = (const float*)d_in[20];
    const float* go  = (const float*)d_in[21];
    const float* bo  = (const float*)d_in[22];

    float* out     = (float*)d_out;
    float* new_xyz = out;                              // (B,M,3)
    float* y       = out + (size_t)BATCH * MPT * 3;    // (B,256,M)

    char* ws = (char*)d_ws;
    size_t off = 0;
    int* idx1 = (int*)(ws + off); off += (size_t)BATCH * MPT * 32 * 4;
    int* idx2 = (int*)(ws + off); off += (size_t)BATCH * MPT * 32 * 4;
    int* idx3 = (int*)(ws + off); off += (size_t)BATCH * MPT * 64 * 4;
    unsigned short* cat  = (unsigned short*)(ws + off); off += (size_t)BATCH * 384 * MPT * 2;
    unsigned short* wb11 = (unsigned short*)(ws + off); off += (size_t)64  * 96  * 2;
    unsigned short* wb12 = (unsigned short*)(ws + off); off += (size_t)128 * 96  * 2;
    unsigned short* wb21 = (unsigned short*)(ws + off); off += (size_t)64  * 96  * 2;
    unsigned short* wb22 = (unsigned short*)(ws + off); off += (size_t)128 * 96  * 2;
    unsigned short* wb31 = (unsigned short*)(ws + off); off += (size_t)96  * 96  * 2;
    unsigned short* wb32 = (unsigned short*)(ws + off); off += (size_t)128 * 96  * 2;
    unsigned short* wbo  = (unsigned short*)(ws + off); off += (size_t)256 * 384 * 2;

    // one-shot bf16 weight images (zero-padded K)
    pad_convert_kernel<<<24, 256, 0, stream>>>(w11, wb11, 64, 67, 96);
    pad_convert_kernel<<<48, 256, 0, stream>>>(w12, wb12, 128, 64, 96);
    pad_convert_kernel<<<24, 256, 0, stream>>>(w21, wb21, 64, 67, 96);
    pad_convert_kernel<<<48, 256, 0, stream>>>(w22, wb22, 128, 64, 96);
    pad_convert_kernel<<<36, 256, 0, stream>>>(w31, wb31, 96, 67, 96);
    pad_convert_kernel<<<48, 256, 0, stream>>>(w32, wb32, 128, 96, 96);
    pad_convert_kernel<<<384, 256, 0, stream>>>(wo, wbo, 256, 384, 384);

    fps_kernel<<<BATCH, 1024, 0, stream>>>(xyz, new_xyz);
    ball_query_kernel<<<(BATCH * MPT) / 8, 256, 0, stream>>>(xyz, new_xyz, idx1, idx2, idx3);

    branch_kernel<<<(BATCH * MPT * 32) / COLS, 256, 0, stream>>>(
        xyz, feats, new_xyz, idx1, 32, 64, wb11, g11, b11, wb12, g12, b12, cat, 0);
    branch_kernel<<<(BATCH * MPT * 32) / COLS, 256, 0, stream>>>(
        xyz, feats, new_xyz, idx2, 32, 64, wb21, g21, b21, wb22, g22, b22, cat, 128);
    branch_kernel<<<(BATCH * MPT * 64) / COLS, 256, 0, stream>>>(
        xyz, feats, new_xyz, idx3, 64, 96, wb31, g31, b31, wb32, g32, b32, cat, 256);

    out_gemm_kernel<<<(BATCH * MPT) / 64, 256, 0, stream>>>(cat, wbo, go, bo, y);
}
